// Qwen3MoeDecoderLayer_20383914787232
// MI455X (gfx1250) — matrix operand order repaired and checked by reading, not compiled, not run on hardware
//
#include <hip/hip_runtime.h>
#include <hip/hip_bf16.h>

// ---------------------------------------------------------------------------
// Qwen3-MoE decoder FFN layer for gfx1250 (MI455X), wave32 + WMMA bf16.
// T=2048 tokens, H=2048 hidden, E=32 experts, I=768 intermediate, top-8.
//
// Pipeline per call:
//   0) zero out + expert counters
//   1) convert fp32 weights -> bf16 in workspace (memory-bound pre-pass;
//      keeps the GEMM inner loop free of conversion VALU ops)
//   2) router: softmax + top-8 + renorm, scatter to per-expert token lists
//   3) fused SwiGLU expert FFN with v_wmma_f32_16x16x32_bf16
// Workspace use: 1 MB control + 3 * E*I*H * 2B = ~303 MB bf16 weights.
// ---------------------------------------------------------------------------

#define TT   2048   // tokens
#define HH   2048   // hidden dim
#define EE   32     // experts
#define II   768    // intermediate dim
#define TOPK 8
#define TM   16     // token tile (WMMA M)

typedef __attribute__((ext_vector_type(16))) __bf16 v16bf;
typedef __attribute__((ext_vector_type(8)))  __bf16 v8bf;
typedef __attribute__((ext_vector_type(8)))  float  v8f;

// --- fp32 -> packed bf16 conversion helpers --------------------------------
static __device__ __forceinline__ unsigned rtne_u(float f) {
    unsigned u = __builtin_bit_cast(unsigned, f);
    return u + 0x7FFFu + ((u >> 16) & 1u);
}

// pack two fp32 into one dword of two bf16 (lo -> [15:0], hi -> [31:16])
static __device__ __forceinline__ unsigned pack2bf(float lo, float hi) {
#if __has_builtin(__builtin_amdgcn_cvt_pk_bf16_f32)
    typedef __attribute__((ext_vector_type(2))) __bf16 v2bf;
    v2bf r = __builtin_amdgcn_cvt_pk_bf16_f32(lo, hi);
    return __builtin_bit_cast(unsigned, r);
#else
    return (rtne_u(lo) >> 16) | (rtne_u(hi) & 0xFFFF0000u);
#endif
}

static __device__ __forceinline__ __bf16 f2bf(float f) {
    unsigned short h = (unsigned short)(rtne_u(f) >> 16);
    return __builtin_bit_cast(__bf16, h);
}

// A-fragment for v_wmma_f32_16x16x32_bf16 from LDS (row-major [16][ld]).
// lane = c*16 + r holds elem i -> K = c*16 + i (matches B fragment)
// => one contiguous 32B run at k0 + c*16.
static __device__ __forceinline__ v16bf load_a_lds(const __bf16* __restrict__ s,
                                                   int ld, int r, int c, int k0) {
    return *(const v16bf*)(s + r * ld + k0 + c * 16);
}

// B-fragment from global bf16. Here lane = c*16 + n holds elem i ->
// B[K = c*16 + i][n]; with B[k][n] = W[n_row][k] this is 16 consecutive
// bf16 = 32 contiguous bytes per lane (32B aligned by construction).
static __device__ __forceinline__ v16bf load_b_bf16(const __bf16* __restrict__ p) {
    return *(const v16bf*)p;
}

// ---------------------------------------------------------------------------
// Kernel 0: zero output + expert counters.
// ---------------------------------------------------------------------------
__global__ __launch_bounds__(256) void zero_kernel(float* __restrict__ out,
                                                   int* __restrict__ cnt) {
    size_t i = (size_t)blockIdx.x * blockDim.x + threadIdx.x;
    const size_t n = (size_t)TT * HH;
    for (size_t idx = i * 4; idx < n; idx += (size_t)gridDim.x * blockDim.x * 4) {
        *(float4*)(out + idx) = make_float4(0.f, 0.f, 0.f, 0.f);
    }
    if (i < EE) cnt[i] = 0;
}

// ---------------------------------------------------------------------------
// Kernel 1: fp32 -> bf16 weight conversion (8 floats / thread, b128 in/out).
// ---------------------------------------------------------------------------
__global__ __launch_bounds__(256) void cvt_kernel(const float* __restrict__ src,
                                                  unsigned* __restrict__ dst,
                                                  size_t n) {
    const size_t base = ((size_t)blockIdx.x * blockDim.x + threadIdx.x) * 8;
    if (base >= n) return;
    const float4 a = *(const float4*)(src + base);
    const float4 b = *(const float4*)(src + base + 4);
    uint4 o;
    o.x = pack2bf(a.x, a.y);
    o.y = pack2bf(a.z, a.w);
    o.z = pack2bf(b.x, b.y);
    o.w = pack2bf(b.z, b.w);
    *(uint4*)(dst + base / 2) = o;
}

// ---------------------------------------------------------------------------
// Kernel 2: router. One wave32 per token; lane == expert (E==32).
// ---------------------------------------------------------------------------
__global__ __launch_bounds__(32) void router_kernel(const float* __restrict__ h,
                                                    const float* __restrict__ gw,
                                                    int*   __restrict__ cnt,
                                                    int*   __restrict__ toks,
                                                    float* __restrict__ scls) {
    const int t = blockIdx.x;
    const int e = threadIdx.x;           // lane == expert
    const float* hr = h  + (size_t)t * HH;
    const float* wr = gw + (size_t)e * HH;

    float acc = 0.f;
    for (int i = 0; i < HH; i += 4) {
        const float4 hv = *(const float4*)(hr + i);
        const float4 wv = *(const float4*)(wr + i);
        acc += hv.x * wv.x + hv.y * wv.y + hv.z * wv.z + hv.w * wv.w;
    }

    // softmax over 32 lanes
    float m = acc;
#pragma unroll
    for (int off = 16; off; off >>= 1) m = fmaxf(m, __shfl_xor(m, off, 32));
    const float ex = __expf(acc - m);
    float s = ex;
#pragma unroll
    for (int off = 16; off; off >>= 1) s += __shfl_xor(s, off, 32);
    const float prob = ex / s;

    // top-8 by iterative wave-wide argmax
    float work = prob;
    bool  sel = false;
    float selsum = 0.f;
#pragma unroll
    for (int k = 0; k < TOPK; ++k) {
        float bv = work;
        int   bi = e;
#pragma unroll
        for (int off = 16; off; off >>= 1) {
            const float ov = __shfl_xor(bv, off, 32);
            const int   oi = __shfl_xor(bi, off, 32);
            if (ov > bv || (ov == bv && oi < bi)) { bv = ov; bi = oi; }
        }
        selsum += bv;
        if (e == bi) { sel = true; work = -1.f; }
    }

    if (sel) {
        const float w = prob / selsum;           // norm_topk_prob=True
        const int pos = atomicAdd(&cnt[e], 1);
        toks[e * TT + pos] = t;
        scls[e * TT + pos] = w;
    }
}

// ---------------------------------------------------------------------------
// Kernel 3: fused SwiGLU expert FFN, expert-major with gathered tokens.
// Block = (expert e, 16-token tile). 256 threads = 8 waves.
// Weights are bf16 (pre-converted): inner loop is ds_load + global b128 +
// v_wmma only.
// ---------------------------------------------------------------------------
__global__ __launch_bounds__(256) void moe_kernel(const float*  __restrict__ hidden,
                                                  const __bf16* __restrict__ wgate,
                                                  const __bf16* __restrict__ wup,
                                                  const __bf16* __restrict__ wdown,
                                                  const int*    __restrict__ cnt,
                                                  const int*    __restrict__ toks,
                                                  const float*  __restrict__ scls,
                                                  float* __restrict__ out) {
    const int tilesPerE = TT / TM;          // 128
    const int e    = blockIdx.x / tilesPerE;
    const int tile = blockIdx.x % tilesPerE;
    const int ne   = cnt[e];
    if (tile * TM >= ne) return;

    extern __shared__ char smem[];
    __bf16* hs  = (__bf16*)smem;            // [TM][HH] bf16   65536 B
    __bf16* as_ = hs + TM * HH;             // [TM][II] bf16   24576 B
    float*  scl = (float*)(as_ + TM * II);  // [TM]
    int*    tok = (int*)(scl + TM);         // [TM]

    const int tid = threadIdx.x;
    if (tid < TM) {
        const int idx = tile * TM + tid;
        if (idx < ne) {
            tok[tid] = toks[e * TT + idx];
            scl[tid] = scls[e * TT + idx];
        } else {                             // pad tail rows: weight 0
            tok[tid] = toks[e * TT + tile * TM];
            scl[tid] = 0.f;
        }
    }
    __syncthreads();

    // Phase 0: gather hidden rows -> LDS bf16 (packed stores)
    for (int base = tid * 4; base < TM * HH; base += 256 * 4) {
        const int j = base / HH, col = base % HH;
        const float4 v = *(const float4*)(hidden + (size_t)tok[j] * HH + col);
        unsigned* d = (unsigned*)(hs + j * HH + col);
        d[0] = pack2bf(v.x, v.y);
        d[1] = pack2bf(v.z, v.w);
    }
    __syncthreads();

    const int wave = tid >> 5;
    const int lane = tid & 31;
    const int c = lane >> 4;                 // lane group (0/1)
    const int n = lane & 15;                 // column (B/C/D) or row (A)

    // ---------------- Phase 1: g = h*Wg^T, u = h*Wu^T ----------------
    v8f accg[6] = {}, accu[6] = {};
    const size_t wb = (size_t)e * II * HH;

    for (int k0 = 0; k0 < HH; k0 += 32) {
        const v16bf afrag = load_a_lds(hs, HH, n, c, k0);
#pragma unroll
        for (int q = 0; q < 6; ++q) {
            const int i0 = (wave * 6 + q) * 16;
            const size_t rowb = wb + (size_t)(i0 + n) * HH + k0 + c * 16;
            const v16bf bg = load_b_bf16(wgate + rowb);
            accg[q] = __builtin_amdgcn_wmma_f32_16x16x32_bf16(
                false, afrag, false, bg, (short)0, accg[q], false, false);
            const v16bf bu = load_b_bf16(wup + rowb);
            accu[q] = __builtin_amdgcn_wmma_f32_16x16x32_bf16(
                false, afrag, false, bu, (short)0, accu[q], false, false);
        }
    }

    // SiLU(g) * u -> LDS activation tile. D elem j of lane c*16+n = row j+8c.
#pragma unroll
    for (int q = 0; q < 6; ++q) {
        const int i0 = (wave * 6 + q) * 16;
#pragma unroll
        for (int j = 0; j < 8; ++j) {
            const float g = accg[q][j];
            const float u = accu[q][j];
            const float a = (g / (1.f + __expf(-g))) * u;
            as_[(j + 8 * c) * II + i0 + n] = f2bf(a);
        }
    }
    __syncthreads();

    // ---------------- Phase 2: y = a * Wd^T, scatter-add ----------------
    int   tokr[8];
    float sclr[8];
#pragma unroll
    for (int j = 0; j < 8; ++j) { tokr[j] = tok[j + 8 * c]; sclr[j] = scl[j + 8 * c]; }

    const size_t db = (size_t)e * HH * II;
    for (int q = 0; q < 16; ++q) {
        const int h0 = (wave + 8 * q) * 16;
        v8f acc = {};
        for (int k0 = 0; k0 < II; k0 += 32) {
            const v16bf afrag = load_a_lds(as_, II, n, c, k0);
            const v16bf bfrag = load_b_bf16(wdown + db + (size_t)(h0 + n) * II + k0 + c * 16);
            acc = __builtin_amdgcn_wmma_f32_16x16x32_bf16(
                false, afrag, false, bfrag, (short)0, acc, false, false);
        }
#pragma unroll
        for (int j = 0; j < 8; ++j) {
            const float v = acc[j] * sclr[j];
            unsafeAtomicAdd(out + (size_t)tokr[j] * HH + h0 + n, v);
        }
    }
}

// ---------------------------------------------------------------------------
extern "C" void kernel_launch(void* const* d_in, const int* in_sizes, int n_in,
                              void* d_out, int out_size, void* d_ws, size_t ws_size,
                              hipStream_t stream) {
    (void)in_sizes; (void)n_in; (void)out_size; (void)ws_size;
    const float* hidden = (const float*)d_in[0];   // [T,H]
    const float* gate_w = (const float*)d_in[1];   // [E,H]
    const float* w_gate = (const float*)d_in[2];   // [E,I,H]
    const float* w_up   = (const float*)d_in[3];   // [E,I,H]
    const float* w_down = (const float*)d_in[4];   // [E,H,I]
    float* out = (float*)d_out;                    // [T,H]

    char* ws = (char*)d_ws;
    int*   cnt  = (int*)ws;                                   // E counters
    int*   toks = (int*)(ws + 256);                           // E*T token ids
    float* scls = (float*)(ws + 256 + (size_t)EE * TT * 4);   // E*T weights

    const size_t WELEMS = (size_t)EE * II * HH;               // 50,331,648
    const size_t WBYTES = WELEMS * 2;                         // bf16 tensor size
    __bf16* wgb = (__bf16*)(ws + (1u << 20));
    __bf16* wub = (__bf16*)((char*)wgb + WBYTES);
    __bf16* wdb = (__bf16*)((char*)wub + WBYTES);

    zero_kernel<<<4096, 256, 0, stream>>>(out, cnt);

    const int cvtBlocks = (int)(WELEMS / (256 * 8));          // 24576
    cvt_kernel<<<cvtBlocks, 256, 0, stream>>>(w_gate, (unsigned*)wgb, WELEMS);
    cvt_kernel<<<cvtBlocks, 256, 0, stream>>>(w_up,   (unsigned*)wub, WELEMS);
    cvt_kernel<<<cvtBlocks, 256, 0, stream>>>(w_down, (unsigned*)wdb, WELEMS);

    router_kernel<<<TT, 32, 0, stream>>>(hidden, gate_w, cnt, toks, scls);

    const size_t smem = (size_t)TM * HH * 2 + (size_t)TM * II * 2 + TM * 4 + TM * 4;
    moe_kernel<<<EE * (TT / TM), 256, smem, stream>>>(hidden, wgb, wub, wdb,
                                                      cnt, toks, scls, out);
}
